// MultiHeadAttention_46196668236416
// MI455X (gfx1250) — matrix operand order repaired and checked by reading, not compiled, not run on hardware
//
#include <hip/hip_runtime.h>

// MHA forward: S=2048, B=2, D=1024, H=16, DK=64.
// Pipeline: 3x projection GEMM (WMMA f16, fused bias+RoPE+scale, f16 out to ws;
//           V stored head-transposed (b,h,dk,s))
//           -> flash attention (async global->LDS staging, WMMA f16, online softmax)
//           -> output GEMM (WMMA f16, fp32 out).
// Workspace: Qh(8MB) | Kh(8MB) | Vt(8MB) | attnO fp32 (16MB) = 40MB.

typedef __attribute__((ext_vector_type(4)))  _Float16 v4h;
typedef __attribute__((ext_vector_type(8)))  _Float16 v8h;
typedef __attribute__((ext_vector_type(16))) _Float16 v16h;
typedef __attribute__((ext_vector_type(8)))  float    v8f;

#define S_LEN 2048
#define BSZ   2
#define DMODEL 1024
#define HEADS 16
#define DK    64

__device__ __forceinline__ v8f wmma_f16(v16h a, v16h b, v8f c) {
  return __builtin_amdgcn_wmma_f32_16x16x32_f16(
      /*neg_a=*/false, a, /*neg_b=*/false, b,
      /*c_mod=*/(short)0, c, /*reuse_a=*/false, /*reuse_b=*/false);
}

// Async global -> LDS 16-byte copy, per lane. LDS offset = low 32 bits of the
// generic pointer to shared memory. Tracked by ASYNCcnt.
__device__ __forceinline__ void async_copy_b128(_Float16* lds_dst, const _Float16* gsrc) {
  const unsigned lds_off = (unsigned)(unsigned long long)lds_dst;
  const unsigned long long ga = (unsigned long long)gsrc;
  asm volatile("global_load_async_to_lds_b128 %0, %1, off"
               :: "v"(lds_off), "v"(ga) : "memory");
}
__device__ __forceinline__ void wait_async0() {
  asm volatile("s_wait_asynccnt 0x0" ::: "memory");
}

// A-fragment 16x32 (MxK), 16-bit: lanes 0-15 row=lane&15 K{h*8..h*8+7,16+h*8..}.
__device__ __forceinline__ v16h frag_a(const _Float16* sm, int r0, int k0, int ld, int lane) {
  const int half = lane >> 4;
  const _Float16* p = sm + (size_t)(r0 + (lane & 15)) * ld + k0 + half * 8;
  v8h lo = *(const v8h*)(p);
  v8h hi = *(const v8h*)(p + 16);
  return __builtin_shufflevector(lo, hi, 0,1,2,3,4,5,6,7,8,9,10,11,12,13,14,15);
}

// B-fragment 32x16 (KxN), 16-bit: lane selects N, K{h*8..h*8+7,16+h*8..} per half.
__device__ __forceinline__ v16h frag_b(const _Float16* sm, int n0, int k0, int ld, int lane) {
  const int half = lane >> 4;
  const _Float16* p = sm + (size_t)(n0 + (lane & 15)) * ld + k0 + half * 8;
  v8h lo = *(const v8h*)(p);
  v8h hi = *(const v8h*)(p + 16);
  return __builtin_shufflevector(lo, hi, 0,1,2,3,4,5,6,7,8,9,10,11,12,13,14,15);
}

// C = A(MxK) * W(NxK)^T + bias.  MODE 0: fp32 out[M,N].
// MODE 1: (RoPE then *oscale), f16 store to (b,h,s,dk).
// MODE 2: *oscale, f16 store head-TRANSPOSED to (b,h,dk,s).
template <int MODE>
__global__ __launch_bounds__(256) void gemm_wmma(
    const float* __restrict__ A, const float* __restrict__ W,
    const float* __restrict__ bias, float* __restrict__ outF,
    _Float16* __restrict__ outH, float oscale)
{
  constexpr int K = DMODEL, N = DMODEL;
  constexpr int LD = 40;
  __shared__ _Float16 As[128 * LD];
  __shared__ _Float16 Bs[128 * LD];

  const int tid  = threadIdx.x;
  const int lane = tid & 31;
  const int wid  = tid >> 5;
  const int tileM = blockIdx.x * 128;
  const int tileN = blockIdx.y * 128;
  const int wm = (wid & 1) * 64;
  const int wn = (wid >> 1) * 32;

  const v8f zacc = {};
  v8f acc[4][2];
#pragma unroll
  for (int mt = 0; mt < 4; ++mt)
#pragma unroll
    for (int nt = 0; nt < 2; ++nt) acc[mt][nt] = zacc;

  for (int k0 = 0; k0 < K; k0 += 32) {
#pragma unroll
    for (int i = 0; i < 4; ++i) {
      const int idx = tid + i * 256;
      const int row = idx >> 3;
      const int cg  = (idx & 7) * 4;
      const float4 a4 = *(const float4*)(A + (size_t)(tileM + row) * K + k0 + cg);
      const float4 w4 = *(const float4*)(W + (size_t)(tileN + row) * K + k0 + cg);
      v4h ah = { (_Float16)a4.x, (_Float16)a4.y, (_Float16)a4.z, (_Float16)a4.w };
      v4h wh = { (_Float16)w4.x, (_Float16)w4.y, (_Float16)w4.z, (_Float16)w4.w };
      *(v4h*)&As[row * LD + cg] = ah;
      *(v4h*)&Bs[row * LD + cg] = wh;
    }
    if (k0 + 32 < K) {
      const int row = tid >> 1, cg = (tid & 1) * 16;
      __builtin_prefetch(A + (size_t)(tileM + row) * K + k0 + 32 + cg, 0, 0);
      __builtin_prefetch(W + (size_t)(tileN + row) * K + k0 + 32 + cg, 0, 0);
    }
    __syncthreads();

    v16h bf0 = frag_b(Bs, wn,      0, LD, lane);
    v16h bf1 = frag_b(Bs, wn + 16, 0, LD, lane);
#pragma unroll
    for (int mt = 0; mt < 4; ++mt) {
      v16h af = frag_a(As, wm + mt * 16, 0, LD, lane);
      acc[mt][0] = wmma_f16(af, bf0, acc[mt][0]);
      acc[mt][1] = wmma_f16(af, bf1, acc[mt][1]);
    }
    __syncthreads();
  }

  const int half = lane >> 4;
  const int cl   = lane & 15;
#pragma unroll
  for (int nt = 0; nt < 2; ++nt) {
    const int col = tileN + wn + nt * 16 + cl;
    const float bv = bias[col];
#pragma unroll
    for (int mt = 0; mt < 4; ++mt) {
#pragma unroll
      for (int i = 0; i < 8; ++i) {
        const int row = tileM + wm + mt * 16 + i + half * 8;
        float v = acc[mt][nt][i] + bv;
        if (MODE == 0) {
          outF[(size_t)row * N + col] = v;
        } else {
          const int s  = row >> 1;             // row = s*B + b, B==2
          const int b  = row & 1;
          const int h  = col >> 6;
          const int dk = col & 63;
          if (MODE == 1) {
            const float pv  = __shfl_xor(v, 1);      // rotate_half pair = col^1
            const float rot = (dk & 1) ? pv : -pv;
            const int   j   = dk & 31;
            const float theta = (float)s * __expf(-0.28782313662425572f * (float)j);
            v = v * __cosf(theta) + rot * __sinf(theta);
          }
          v *= oscale;
          if (MODE == 1)
            outH[((size_t)(b * HEADS + h) * S_LEN + s) * DK + dk] = (_Float16)v;
          else                                  // MODE 2: (b,h,dk,s)
            outH[((size_t)(b * HEADS + h) * DK + dk) * S_LEN + s] = (_Float16)v;
        }
      }
    }
  }
}

// Flash attention: grid (S/128, B*H), 256 threads (8 waves).
// Wave w owns q-rows [qBlock + 16w, +16). Scale folded into Q. Full softmax.
__global__ __launch_bounds__(256) void attn_wmma(
    const _Float16* __restrict__ Qh, const _Float16* __restrict__ Kh,
    const _Float16* __restrict__ Vt, float* __restrict__ O)
{
  constexpr int LD = 72;
  __shared__ _Float16 Ks [64 * LD];            // [key][dk]
  __shared__ _Float16 Vts[64 * LD];            // [dk][key]
  __shared__ _Float16 Ps [8][16 * LD];         // per-wave P tile [qrow][key]

  const int tid  = threadIdx.x;
  const int lane = tid & 31;
  const int wid  = tid >> 5;
  const int half = lane >> 4;
  const int cl   = lane & 15;
  const int qBlock = blockIdx.x * 128;
  const int z = blockIdx.y;                    // b*H + h
  const int b = z >> 4, h = z & 15;

  const _Float16* Qp = Qh + ((size_t)z * S_LEN + qBlock + wid * 16) * DK;
  const _Float16* Kp = Kh + (size_t)z * S_LEN * DK;     // [s][dk]
  const _Float16* Vp = Vt + (size_t)z * DK * S_LEN;     // [dk][s]

  v16h qf[2];
#pragma unroll
  for (int kc = 0; kc < 2; ++kc) qf[kc] = frag_a(Qp, 0, kc * 32, DK, lane);

  const v8f zacc = {};
  v8f o_acc[4];
#pragma unroll
  for (int nt = 0; nt < 4; ++nt) o_acc[nt] = zacc;
  float m_run[8], l_run[8];
#pragma unroll
  for (int r = 0; r < 8; ++r) { m_run[r] = -3.0e38f; l_run[r] = 0.0f; }

  for (int kb = 0; kb < S_LEN; kb += 64) {
    __syncthreads();                           // prior tiles fully consumed
    // Async-stage K [key][dk] and Vt [dk][key] 64x64 f16 tiles (16B per lane).
#pragma unroll
    for (int i = 0; i < 2; ++i) {
      const int idx = tid + i * 256;           // 0..511
      const int row = idx >> 3;
      const int cg  = (idx & 7) * 8;
      async_copy_b128(&Ks[row * LD + cg],  Kp + (size_t)(kb + row) * DK + cg);
      async_copy_b128(&Vts[row * LD + cg], Vp + (size_t)row * S_LEN + kb + cg);
    }
    wait_async0();
    __syncthreads();

    // Scores: 16(q) x 64(keys) per wave.
    v8f sc[4];
#pragma unroll
    for (int nt = 0; nt < 4; ++nt) {
      sc[nt] = zacc;
#pragma unroll
      for (int kc = 0; kc < 2; ++kc)
        sc[nt] = wmma_f16(qf[kc], frag_b(Ks, nt * 16, kc * 32, LD, lane), sc[nt]);
    }

    // Online softmax (rows r+8*half; reductions within each 16-lane half).
    float p_[4][8];
#pragma unroll
    for (int r = 0; r < 8; ++r) {
      float mx = fmaxf(fmaxf(sc[0][r], sc[1][r]), fmaxf(sc[2][r], sc[3][r]));
#pragma unroll
      for (int off = 8; off >= 1; off >>= 1) mx = fmaxf(mx, __shfl_xor(mx, off));
      const float mnew  = fmaxf(m_run[r], mx);
      const float alpha = __expf(m_run[r] - mnew);
      m_run[r] = mnew;
      float rs = 0.0f;
#pragma unroll
      for (int nt = 0; nt < 4; ++nt) {
        const float e = __expf(sc[nt][r] - mnew);
        p_[nt][r] = e; rs += e;
      }
#pragma unroll
      for (int off = 8; off >= 1; off >>= 1) rs += __shfl_xor(rs, off);
      l_run[r] = l_run[r] * alpha + rs;
#pragma unroll
      for (int nt = 0; nt < 4; ++nt) o_acc[nt][r] *= alpha;
    }

    // P (C layout) -> per-wave LDS row-major [qrow][key] for A-fragments.
    _Float16* myP = &Ps[wid][0];
#pragma unroll
    for (int nt = 0; nt < 4; ++nt)
#pragma unroll
      for (int r = 0; r < 8; ++r)
        myP[(r + half * 8) * LD + nt * 16 + cl] = (_Float16)p_[nt][r];
    asm volatile("s_wait_dscnt 0x0" ::: "memory");   // in-order DS: stores visible

    // O += P @ V.
#pragma unroll
    for (int kc = 0; kc < 2; ++kc) {
      v16h pa = frag_a(myP, 0, kc * 32, LD, lane);
#pragma unroll
      for (int nt = 0; nt < 4; ++nt)
        o_acc[nt] = wmma_f16(pa, frag_b(Vts, nt * 16, kc * 32, LD, lane), o_acc[nt]);
    }
  }

  // Normalize and store fp32 to (s, b, D) with D-col = h*64 + dk.
#pragma unroll
  for (int nt = 0; nt < 4; ++nt) {
    const int dk = nt * 16 + cl;
#pragma unroll
    for (int r = 0; r < 8; ++r) {
      const int s = qBlock + wid * 16 + r + half * 8;
      const float val = o_acc[nt][r] / l_run[r];
      O[((size_t)s * BSZ + b) * DMODEL + h * DK + dk] = val;
    }
  }
}

extern "C" void kernel_launch(void* const* d_in, const int* in_sizes, int n_in,
                              void* d_out, int out_size, void* d_ws, size_t ws_size,
                              hipStream_t stream) {
  (void)in_sizes; (void)n_in; (void)out_size; (void)ws_size;
  const float* query = (const float*)d_in[0];
  const float* key   = (const float*)d_in[1];
  const float* value = (const float*)d_in[2];
  const float* wq = (const float*)d_in[3];
  const float* bq = (const float*)d_in[4];
  const float* wk = (const float*)d_in[5];
  const float* bk = (const float*)d_in[6];
  const float* wv = (const float*)d_in[7];
  const float* bv = (const float*)d_in[8];
  const float* wo = (const float*)d_in[9];
  const float* bo = (const float*)d_in[10];
  float* out = (float*)d_out;

  char* ws = (char*)d_ws;
  _Float16* Qh = (_Float16*)(ws);
  _Float16* Kh = (_Float16*)(ws + (size_t)8  * 1024 * 1024);
  _Float16* Vh = (_Float16*)(ws + (size_t)16 * 1024 * 1024);
  float*    Ao = (float*)   (ws + (size_t)24 * 1024 * 1024);

  const dim3 gG(32, 8), bG(256);
  gemm_wmma<1><<<gG, bG, 0, stream>>>(query, wq, bq, nullptr, Qh, 0.125f); // 1/sqrt(DK)
  gemm_wmma<1><<<gG, bG, 0, stream>>>(key,   wk, bk, nullptr, Kh, 1.0f);
  gemm_wmma<2><<<gG, bG, 0, stream>>>(value, wv, bv, nullptr, Vh, 1.0f);
  attn_wmma<<<dim3(S_LEN / 128, BSZ * HEADS), 256, 0, stream>>>(Qh, Kh, Vh, Ao);
  gemm_wmma<0><<<gG, bG, 0, stream>>>(Ao, wo, bo, out, nullptr, 1.0f);
}
